// MultiHeadSelfAttention_3238405341415
// MI455X (gfx1250) — matrix operand order repaired and checked by reading, not compiled, not run on hardware
//
#include <hip/hip_runtime.h>
#include <hip/hip_bf16.h>

typedef __attribute__((ext_vector_type(16))) __bf16 bf16x16;
typedef __attribute__((ext_vector_type(8)))  __bf16 bf16x8;
typedef __attribute__((ext_vector_type(8)))  float  f32x8;
typedef __attribute__((ext_vector_type(4)))  int    v4i;
typedef __attribute__((address_space(1))) v4i gbl_v4i;   // global (AS1)
typedef __attribute__((address_space(3))) v4i lds_v4i;   // LDS (AS3)

#define DIM 768
#define HEADS 12
#define DHEAD 64
#define SEQ 2048
#define BATCH 4
#define NROWS (BATCH * SEQ)      // 8192
#define QKVN (3 * DIM)           // 2304
#define FACTOR 0.125f            // 64^-0.5

// ---------------- CDNA5 async global->LDS (ASYNCcnt-tracked DMA) ----------------
#if __has_builtin(__builtin_amdgcn_global_load_async_to_lds_b128)
#define HAS_ASYNC 1
__device__ __forceinline__ void async_cp16(const __bf16* g, __bf16* s) {
    __builtin_amdgcn_global_load_async_to_lds_b128(
        (gbl_v4i*)g, (lds_v4i*)s, 0, 0);
}
#else
#define HAS_ASYNC 0
__device__ __forceinline__ void async_cp16(const __bf16* g, __bf16* s) {
    *(uint4*)s = *(const uint4*)g;   // sync fallback: global->VGPR->ds_store
}
#endif

__device__ __forceinline__ void wait_async0() {
#if HAS_ASYNC
#if __has_builtin(__builtin_amdgcn_s_wait_asynccnt)
    __builtin_amdgcn_s_wait_asynccnt(0);
#else
    asm volatile("s_wait_asynccnt 0x0" ::: "memory");
#endif
#endif
}

// ---------------- fragment loaders (16-bit WMMA operands; contiguous k order, same for A and B) ----------------
// A-matrix 16x32 bf16: lane L holds row M=L%16.
//   j=0..15 -> K = j + 16*(L>=16)
// caller passes p already offset by (L>=16)*16 elements; read 16 contiguous.
__device__ __forceinline__ bf16x16 load_frag_a(const __bf16* p) {
    bf16x16 r;
    *(bf16x8*)&r       = *(const bf16x8*)p;
    *((bf16x8*)&r + 1) = *(const bf16x8*)(p + 8);
    return r;
}
// B-matrix 32x16 bf16: lane L holds col N=L%16; j=0..15 -> K = j + 16*(L>=16).
// caller passes p offset by (L>=16)*16 elements; 16 contiguous K values.
__device__ __forceinline__ bf16x16 load_frag_b(const __bf16* p) {
    bf16x16 r;
    *(bf16x8*)&r       = *(const bf16x8*)p;
    *((bf16x8*)&r + 1) = *(const bf16x8*)(p + 8);
    return r;
}

// ---------------- prep kernels ----------------
__global__ void convert_x_kernel(const float* __restrict__ x, __bf16* __restrict__ xb, int n) {
    int i = blockIdx.x * 256 + threadIdx.x;
    if (i < n) xb[i] = (__bf16)x[i];
}

// W_qkv [768][2304] f32, einops '(d k h)' columns -> store transposed-permuted
// WqkvT[n][r] with n = kidx*768 + h*64 + d  mapping to col = d*36 + kidx*12 + h
__global__ void prep_wqkv_kernel(const float* __restrict__ W, __bf16* __restrict__ Wt) {
    int i = blockIdx.x * 256 + threadIdx.x;
    if (i >= QKVN * DIM) return;
    int n = i / DIM, r = i % DIM;
    int kidx = n / DIM;
    int rem  = n % DIM;
    int h = rem / DHEAD, d = rem % DHEAD;
    int col = d * (3 * HEADS) + kidx * HEADS + h;
    Wt[(size_t)n * DIM + r] = (__bf16)W[(size_t)r * QKVN + col];
}

// W_out [768][768] f32 -> transposed bf16 [N][K]
__global__ void prep_wout_kernel(const float* __restrict__ W, __bf16* __restrict__ Wt) {
    int i = blockIdx.x * 256 + threadIdx.x;
    if (i >= DIM * DIM) return;
    int n = i / DIM, r = i % DIM;
    Wt[(size_t)n * DIM + r] = (__bf16)W[(size_t)r * DIM + n];
}

// ---------------- bf16 GEMM: C[M][N] = A[M][K] * Bt[N][K]^T ----------------
// block 256 thr = 8 waves (4x2), tile 128(M) x 128(N), K-step 64,
// double-buffered LDS filled by async DMA overlapped with WMMA compute.
#define GLDT 72   // padded LDS stride (elems), 144B rows
template <typename OutT>
__global__ __launch_bounds__(256) void gemm_bf16_kernel(
    const __bf16* __restrict__ A, const __bf16* __restrict__ Bt,
    OutT* __restrict__ C, int M, int N, int K)
{
    __shared__ __align__(16) __bf16 As[2][128 * GLDT];
    __shared__ __align__(16) __bf16 Bs[2][128 * GLDT];

    const int tid  = threadIdx.x;
    const int lane = tid & 31;
    const int wave = tid >> 5;
    const int wm   = wave >> 1;                   // 0..3
    const int wn   = wave & 1;                    // 0..1
    const int lmod = lane & 15;
    const int lhalf = lane >> 4;
    const int bm = blockIdx.y * 128;
    const int bn = blockIdx.x * 128;

    f32x8 acc[2][4] = {};

    // global->LDS DMA mapping: each thread moves 32 bf16 (64B) of A and of B
    const int row  = tid >> 1;                    // 0..127
    const int koff = (tid & 1) * 32;              // 0 or 32
    const __bf16* gA = A  + (size_t)(bm + row) * K + koff;
    const __bf16* gB = Bt + (size_t)(bn + row) * K + koff;
    const int soff = row * GLDT + koff;

    const int nK = K / 64;
    // prologue: DMA tile 0 into buffer 0
#pragma unroll
    for (int i = 0; i < 4; ++i) {
        async_cp16(gA + i * 8, As[0] + soff + i * 8);
        async_cp16(gB + i * 8, Bs[0] + soff + i * 8);
    }

    for (int it = 0; it < nK; ++it) {
        wait_async0();            // this wave's DMA for tile `it` landed in LDS
        __syncthreads();          // everyone's landed; prev buffer fully read
        if (it + 1 < nK) {
            const __bf16* ga = gA + (it + 1) * 64;
            const __bf16* gb = gB + (it + 1) * 64;
            __bf16* sa = As[(it + 1) & 1] + soff;
            __bf16* sb = Bs[(it + 1) & 1] + soff;
#pragma unroll
            for (int i = 0; i < 4; ++i) {
                async_cp16(ga + i * 8, sa + i * 8);
                async_cp16(gb + i * 8, sb + i * 8);
            }
        }
        const __bf16* rA = As[it & 1] + (wm * 32 + lmod) * GLDT + lhalf * 16;
        const __bf16* rB = Bs[it & 1] + (wn * 64 + lmod) * GLDT + lhalf * 16;
#pragma unroll
        for (int c = 0; c < 2; ++c) {             // two K=32 chunks
            bf16x16 af[2], bfr[4];
#pragma unroll
            for (int tm = 0; tm < 2; ++tm) af[tm] = load_frag_a(rA + tm * 16 * GLDT + c * 32);
#pragma unroll
            for (int tn = 0; tn < 4; ++tn) bfr[tn] = load_frag_b(rB + tn * 16 * GLDT + c * 32);
#pragma unroll
            for (int tm = 0; tm < 2; ++tm)
#pragma unroll
                for (int tn = 0; tn < 4; ++tn)
                    acc[tm][tn] = __builtin_amdgcn_wmma_f32_16x16x32_bf16(
                        false, af[tm], false, bfr[tn], (short)0, acc[tm][tn], false, false);
        }
    }

    // epilogue: C element (v,lane) = (M = v + 8*lhalf, N = lmod) per 16x16 tile
#pragma unroll
    for (int tm = 0; tm < 2; ++tm) {
#pragma unroll
        for (int tn = 0; tn < 4; ++tn) {
#pragma unroll
            for (int v = 0; v < 8; ++v) {
                int r = bm + wm * 32 + tm * 16 + v + 8 * lhalf;
                int c = bn + wn * 64 + tn * 16 + lmod;
                C[(size_t)r * N + c] = (OutT)acc[tm][tn][v];
            }
        }
    }
}

// ---------------- flash attention ----------------
// QKV bf16 [B*T][2304] laid out as [q|k|v] x [h][d] contiguous per row.
// grid: (T/128, B*HEADS); block 256 = 8 waves, each wave owns 16 query rows.
__global__ __launch_bounds__(256) void flash_attn_kernel(
    const __bf16* __restrict__ QKV, __bf16* __restrict__ Oattn)
{
    const int LDK = 64 + 8;                       // padded LDS stride (144B rows)
    __shared__ __align__(16) __bf16 Ks[64 * LDK]; // K tile  [j][d]
    __shared__ __align__(16) __bf16 Vt[64 * LDK]; // V tile transposed [d][j]
    __shared__ __align__(16) __bf16 Pw[8][16 * LDK]; // per-wave P tile [m][j]

    const int bh = blockIdx.y;
    const int b = bh / HEADS, h = bh % HEADS;
    const int lane  = threadIdx.x & 31;
    const int wave  = threadIdx.x >> 5;
    const int lmod  = lane & 15;
    const int lhalf = lane >> 4;
    const size_t rowbase = (size_t)b * SEQ;
    const int qrow0 = blockIdx.x * 128 + wave * 16;

    // load Q fragments once (2 k-chunks of 32 over d=64), kept in registers
    const __bf16* Qg = QKV + (rowbase + qrow0 + lmod) * QKVN + h * DHEAD;
    bf16x16 qf[2];
#pragma unroll
    for (int c = 0; c < 2; ++c)
        qf[c] = load_frag_a(Qg + c * 32 + lhalf * 16);

    float mrow[8], lrow[8];
#pragma unroll
    for (int v = 0; v < 8; ++v) { mrow[v] = -3.0e38f; lrow[v] = 0.f; }
    f32x8 acc[4] = {};

    // cooperative tile-load mapping
    const int jr = threadIdx.x >> 2;              // 0..63
    const int dc = (threadIdx.x & 3) * 16;        // 0,16,32,48

    for (int jt = 0; jt < SEQ; jt += 64) {
        __syncthreads();
        {   // K tile via async DMA: Ks[j][d]
            const __bf16* g = QKV + (rowbase + jt + jr) * QKVN + DIM + h * DHEAD + dc;
            __bf16* s = Ks + jr * LDK + dc;
            async_cp16(g, s);
            async_cp16(g + 8, s + 8);
        }
        {   // V tile transposed through registers: Vt[d][j]
            const __bf16* g = QKV + (rowbase + jt + jr) * QKVN + 2 * DIM + h * DHEAD + dc;
            union { uint4 u[2]; __bf16 e[16]; } t;
            t.u[0] = *(const uint4*)g;
            t.u[1] = *(const uint4*)(g + 8);
#pragma unroll
            for (int i = 0; i < 16; ++i) Vt[(dc + i) * LDK + jr] = t.e[i];
        }
        wait_async0();
        __syncthreads();

        // S = Q * K^T  (M=16 rows, N=64 cols, K=64) -> 4 n-tiles x 2 k-chunks
        f32x8 S[4];
#pragma unroll
        for (int nt = 0; nt < 4; ++nt) {
            f32x8 s = {};
#pragma unroll
            for (int c = 0; c < 2; ++c) {
                bf16x16 kb = load_frag_b(Ks + (nt * 16 + lmod) * LDK + c * 32 + lhalf * 16);
                s = __builtin_amdgcn_wmma_f32_16x16x32_bf16(
                        false, qf[c], false, kb, (short)0, s, false, false);
            }
            S[nt] = s;
        }

        // online softmax (rows live in fixed 16-lane halves; reduce with xor 1..8)
#pragma unroll
        for (int v = 0; v < 8; ++v) {
            float tmax = -3.0e38f;
#pragma unroll
            for (int nt = 0; nt < 4; ++nt) {
                S[nt][v] *= FACTOR;
                tmax = fmaxf(tmax, S[nt][v]);
            }
#pragma unroll
            for (int m = 1; m < 16; m <<= 1) tmax = fmaxf(tmax, __shfl_xor(tmax, m, 32));
            float mnew  = fmaxf(mrow[v], tmax);
            float alpha = __expf(mrow[v] - mnew);
            mrow[v] = mnew;
            float rsum = 0.f;
#pragma unroll
            for (int nt = 0; nt < 4; ++nt) {
                float p = __expf(S[nt][v] - mnew);
                S[nt][v] = p;
                rsum += p;
            }
#pragma unroll
            for (int m = 1; m < 16; m <<= 1) rsum += __shfl_xor(rsum, m, 32);
            lrow[v] = lrow[v] * alpha + rsum;
#pragma unroll
            for (int dt = 0; dt < 4; ++dt) acc[dt][v] *= alpha;
        }

        // C-layout -> A-layout via per-wave LDS round trip
        __bf16* pw = &Pw[wave][0];
#pragma unroll
        for (int nt = 0; nt < 4; ++nt)
#pragma unroll
            for (int v = 0; v < 8; ++v)
                pw[(v + 8 * lhalf) * LDK + nt * 16 + lmod] = (__bf16)S[nt][v];

        // acc += P * V  (N=d 64, K=j 64)
#pragma unroll
        for (int c = 0; c < 2; ++c) {
            bf16x16 pf = load_frag_a(pw + lmod * LDK + c * 32 + lhalf * 16);
#pragma unroll
            for (int dt = 0; dt < 4; ++dt) {
                bf16x16 vf = load_frag_b(Vt + (dt * 16 + lmod) * LDK + c * 32 + lhalf * 16);
                acc[dt] = __builtin_amdgcn_wmma_f32_16x16x32_bf16(
                              false, pf, false, vf, (short)0, acc[dt], false, false);
            }
        }
    }

    // epilogue: out = acc / l -> bf16 Oattn[b*T + t][h*64 + d]
#pragma unroll
    for (int v = 0; v < 8; ++v) {
        float inv = 1.f / lrow[v];
        int r = qrow0 + v + 8 * lhalf;
#pragma unroll
        for (int dt = 0; dt < 4; ++dt) {
            float o = acc[dt][v] * inv;
            Oattn[(rowbase + r) * DIM + h * DHEAD + dt * 16 + lmod] = (__bf16)o;
        }
    }
}

// ---------------- host launcher ----------------
extern "C" void kernel_launch(void* const* d_in, const int* in_sizes, int n_in,
                              void* d_out, int out_size, void* d_ws, size_t ws_size,
                              hipStream_t stream) {
    (void)in_sizes; (void)n_in; (void)out_size; (void)ws_size;
    const float* x    = (const float*)d_in[0];
    const float* Wqkv = (const float*)d_in[1];
    const float* Wout = (const float*)d_in[2];
    float* out = (float*)d_out;

    __bf16* xb    = (__bf16*)d_ws;                         // [8192][768]
    __bf16* wqkvT = xb    + (size_t)NROWS * DIM;           // [2304][768]
    __bf16* woutT = wqkvT + (size_t)QKVN * DIM;            // [768][768]
    __bf16* qkv   = woutT + (size_t)DIM * DIM;             // [8192][2304]
    __bf16* attn  = qkv   + (size_t)NROWS * QKVN;          // [8192][768]

    convert_x_kernel<<<(NROWS * DIM + 255) / 256, 256, 0, stream>>>(x, xb, NROWS * DIM);
    prep_wqkv_kernel<<<(QKVN * DIM + 255) / 256, 256, 0, stream>>>(Wqkv, wqkvT);
    prep_wout_kernel<<<(DIM * DIM + 255) / 256, 256, 0, stream>>>(Wout, woutT);

    gemm_bf16_kernel<__bf16><<<dim3(QKVN / 128, NROWS / 128), 256, 0, stream>>>(
        xb, wqkvT, qkv, NROWS, QKVN, DIM);

    flash_attn_kernel<<<dim3(SEQ / 128, BATCH * HEADS), 256, 0, stream>>>(qkv, attn);

    gemm_bf16_kernel<float><<<dim3(DIM / 128, NROWS / 128), 256, 0, stream>>>(
        attn, woutT, out, NROWS, DIM, DIM);
}
